// TransformerLayerWithAttention_37864431682200
// MI455X (gfx1250) — compile-verified, matrix operand order checked by reading, not run on hardware
//
#include <hip/hip_runtime.h>
#include <hip/hip_bf16.h>
#include <math.h>

// ---------------------------------------------------------------------------
// CDNA5 (gfx1250) transformer encoder layer, bf16-WMMA GEMM core:
//   qkv -> scores(masked) -> softmax -> attn@V -> out-proj+res -> LN1
//       -> FFN1(GELU) -> FFN2+res -> LN2
// 128x128x32 tiles, double-buffered LDS, fp32->bf16 via native v_cvt_pk_bf16,
// staging split into batched-load + convert/store phases to keep multiple
// global_load_b128 in flight.
// ---------------------------------------------------------------------------

typedef __attribute__((ext_vector_type(16))) __bf16 v16bf;
typedef __attribute__((ext_vector_type(2)))  __bf16 v2bf;
typedef __attribute__((ext_vector_type(8)))  float  v8f;

#define BM 128
#define BN 128
#define BK 32
#define LDSP 40   // padded LDS row stride in halves (80B: keeps 16B alignment)

// single fp32 -> bf16 via native fptrunc (RNE)
__device__ __forceinline__ unsigned short f2bf(float f) {
    union { __bf16 h; unsigned short s; } c;
    c.h = (__bf16)f;
    return c.s;
}

// pack two fp32 into one dword of bf16x2 via native fptrunc
__device__ __forceinline__ unsigned pk2bf(float a, float b) {
    union { v2bf v; unsigned u; } cv;
    cv.v.x = (__bf16)a;
    cv.v.y = (__bf16)b;
    return cv.u;
}

union FragBF {
    uint4 q[2];   // two 16B chunks of 8 halves each
    v16bf v;
};

constexpr int EPI_BIAS_QSCALE = 0;  // + bias; cols < 768 scaled by 0.125 (Q)
constexpr int EPI_MASK        = 1;  // key-padding mask -> -inf
constexpr int EPI_NONE        = 2;
constexpr int EPI_BIAS_RES    = 3;  // + bias + residual
constexpr int EPI_BIAS_GELU   = 4;  // + bias, exact GELU

// ---- staging helpers: global fp32 tile -> LDS bf16 (N-major for B) ----
// Phase 1 issues all global loads (keeps them in flight), phase 2 converts+stores.

template<bool GUARD>
__device__ __forceinline__ void stage_rowmajor(
    unsigned short* __restrict__ S, const float* __restrict__ G, int ldg,
    int r0, int k0, int Rmax, int Kmax, int tid)
{
    float4 vals[4];
#pragma unroll
    for (int it = 0; it < 4; ++it) {
        int lin = it * 256 + tid;
        int r   = lin >> 3;
        int c4  = (lin & 7) << 2;
        if (GUARD) {
            vals[it] = make_float4(0.f, 0.f, 0.f, 0.f);
            if (r0 + r < Rmax && k0 + c4 < Kmax)
                vals[it] = *(const float4*)(G + (long long)(r0 + r) * ldg + (k0 + c4));
        } else {
            vals[it] = *(const float4*)(G + (long long)(r0 + r) * ldg + (k0 + c4));
        }
    }
#pragma unroll
    for (int it = 0; it < 4; ++it) {
        int lin = it * 256 + tid;
        int r   = lin >> 3;
        int c4  = (lin & 7) << 2;
        unsigned* d = (unsigned*)&S[r * LDSP + c4];
        d[0] = pk2bf(vals[it].x, vals[it].y);
        d[1] = pk2bf(vals[it].z, vals[it].w);
    }
}

template<bool GUARD>
__device__ __forceinline__ void stage_transpose(
    unsigned short* __restrict__ S, const float* __restrict__ G, int ldg,
    int n0, int k0, int Nmax, int Kmax, int tid)
{
    // G is [K,N] row-major; store S[n][k]
    float4 vals[4];
#pragma unroll
    for (int it = 0; it < 4; ++it) {
        int lin = it * 256 + tid;
        int kr  = lin >> 5;         // k within tile: 0..31
        int n4  = (lin & 31) << 2;  // n within tile, step 4
        if (GUARD) {
            vals[it] = make_float4(0.f, 0.f, 0.f, 0.f);
            if (k0 + kr < Kmax && n0 + n4 < Nmax)
                vals[it] = *(const float4*)(G + (long long)(k0 + kr) * ldg + (n0 + n4));
        } else {
            vals[it] = *(const float4*)(G + (long long)(k0 + kr) * ldg + (n0 + n4));
        }
    }
#pragma unroll
    for (int it = 0; it < 4; ++it) {
        int lin = it * 256 + tid;
        int kr  = lin >> 5;
        int n4  = (lin & 31) << 2;
        S[(n4 + 0) * LDSP + kr] = f2bf(vals[it].x);
        S[(n4 + 1) * LDSP + kr] = f2bf(vals[it].y);
        S[(n4 + 2) * LDSP + kr] = f2bf(vals[it].z);
        S[(n4 + 3) * LDSP + kr] = f2bf(vals[it].w);
    }
}

// C[M,N] = A[M,K] * op(B) ; BT=true: B stored [N,K] (A*B^T), else B stored [K,N]
// batch offset for X = (z/sub)*xOuter + (z%sub)*xInner
template<int EPI, bool BT, bool GUARD>
__global__ __launch_bounds__(256) void gemm_bf16_wmma(
    const float* __restrict__ A, int lda, long long aOuter, long long aInner,
    const float* __restrict__ B, int ldb, long long bOuter, long long bInner,
    float* __restrict__ C, int ldc, long long cOuter, long long cInner,
    int M, int N, int K, int sub,
    const float* __restrict__ bias,
    const float* __restrict__ resid, int ldr,
    const unsigned char* __restrict__ mask)
{
    __shared__ __align__(16) unsigned short As[2][BM * LDSP];
    __shared__ __align__(16) unsigned short Bs[2][BN * LDSP];

    const int tid  = threadIdx.x;
    const int wave = tid >> 5;
    const int lane = tid & 31;
    const int lrow = lane & 15;
    const int hi   = lane >> 4;
    const int wm   = wave & 3;   // 4 waves along M: 32 rows each
    const int wn   = wave >> 2;  // 2 waves along N: 64 cols each

    const int m0 = blockIdx.y * BM;
    const int n0 = blockIdx.x * BN;
    const int z  = blockIdx.z;
    const int zo = z / sub;
    const int zi = z - zo * sub;

    A += (long long)zo * aOuter + (long long)zi * aInner;
    B += (long long)zo * bOuter + (long long)zi * bInner;
    C += (long long)zo * cOuter + (long long)zi * cInner;

    v8f acc[2][4];
    const v8f zero8 = {0.f, 0.f, 0.f, 0.f, 0.f, 0.f, 0.f, 0.f};
#pragma unroll
    for (int mt = 0; mt < 2; ++mt)
#pragma unroll
        for (int nt = 0; nt < 4; ++nt) acc[mt][nt] = zero8;

    // prologue: stage first tile into buffer 0
    stage_rowmajor<GUARD>(As[0], A, lda, m0, 0, M, K, tid);
    if (BT) stage_rowmajor<GUARD>(Bs[0], B, ldb, n0, 0, N, K, tid);
    else    stage_transpose<GUARD>(Bs[0], B, ldb, n0, 0, N, K, tid);
    __syncthreads();

    int cur = 0;
    for (int kt = 0; kt < K; kt += BK) {
        int nxt = cur ^ 1;
        if (kt + BK < K) {   // prefetch next tile into the other buffer
            stage_rowmajor<GUARD>(As[nxt], A, lda, m0, kt + BK, M, K, tid);
            if (BT) stage_rowmajor<GUARD>(Bs[nxt], B, ldb, n0, kt + BK, N, K, tid);
            else    stage_transpose<GUARD>(Bs[nxt], B, ldb, n0, kt + BK, N, K, tid);
        }

        // ---- fragments per ISA 16-bit A layout; B staged symmetrically ----
        FragBF fa[2], fb[4];
#pragma unroll
        for (int mt = 0; mt < 2; ++mt) {
            int row = (wm * 32 + mt * 16 + lrow) * LDSP;
            fa[mt].q[0] = *(const uint4*)&As[cur][row + hi * 8];       // K 0..7 (+8*hi)
            fa[mt].q[1] = *(const uint4*)&As[cur][row + 16 + hi * 8];  // K 16..23 (+8*hi)
        }
#pragma unroll
        for (int nt = 0; nt < 4; ++nt) {
            int row = (wn * 64 + nt * 16 + lrow) * LDSP;
            fb[nt].q[0] = *(const uint4*)&Bs[cur][row + hi * 8];
            fb[nt].q[1] = *(const uint4*)&Bs[cur][row + 16 + hi * 8];
        }
#pragma unroll
        for (int mt = 0; mt < 2; ++mt)
#pragma unroll
            for (int nt = 0; nt < 4; ++nt)
                acc[mt][nt] = __builtin_amdgcn_wmma_f32_16x16x32_bf16(
                    false, fa[mt].v, false, fb[nt].v,
                    (short)0, acc[mt][nt], false, false);
        __syncthreads();
        cur = nxt;
    }

    // ---- epilogue + store (C/D layout: m = r + 8*hi, n = lane%16) ----
#pragma unroll
    for (int mt = 0; mt < 2; ++mt) {
#pragma unroll
        for (int nt = 0; nt < 4; ++nt) {
            int gn = n0 + wn * 64 + nt * 16 + lrow;
#pragma unroll
            for (int r = 0; r < 8; ++r) {
                int gm = m0 + wm * 32 + mt * 16 + r + hi * 8;
                if (!GUARD || (gm < M && gn < N)) {
                    float v = acc[mt][nt][r];
                    if (EPI == EPI_BIAS_QSCALE) {
                        v += bias[gn];
                        if (gn < 768) v *= 0.125f;      // 1/sqrt(64) folded into Q
                    } else if (EPI == EPI_MASK) {
                        if (mask[zo * 512 + gn]) v = -INFINITY;
                    } else if (EPI == EPI_BIAS_RES) {
                        v += bias[gn] + resid[(long long)gm * ldr + gn];
                    } else if (EPI == EPI_BIAS_GELU) {
                        v += bias[gn];
                        v = 0.5f * v * (1.0f + erff(v * 0.70710678118654752f));
                    }
                    C[(long long)gm * ldc + gn] = v;
                }
            }
        }
    }
}

// one wave per row of 512; 8 rows per block
__global__ __launch_bounds__(256) void softmax512(float* __restrict__ attn, int rows)
{
    int row  = blockIdx.x * 8 + (threadIdx.x >> 5);
    int lane = threadIdx.x & 31;
    if (row >= rows) return;
    float* p = attn + (long long)row * 512;
    float v[16];
    float mx = -INFINITY;
#pragma unroll
    for (int i = 0; i < 16; ++i) { v[i] = p[lane + i * 32]; mx = fmaxf(mx, v[i]); }
#pragma unroll
    for (int o = 16; o > 0; o >>= 1) mx = fmaxf(mx, __shfl_xor(mx, o, 32));
    float s = 0.f;
#pragma unroll
    for (int i = 0; i < 16; ++i) { v[i] = expf(v[i] - mx); s += v[i]; }
#pragma unroll
    for (int o = 16; o > 0; o >>= 1) s += __shfl_xor(s, o, 32);
    float inv = 1.0f / s;
#pragma unroll
    for (int i = 0; i < 16; ++i) p[lane + i * 32] = v[i] * inv;
}

// one 256-thread block per row of 768
__global__ __launch_bounds__(256) void layernorm768(
    const float* __restrict__ in, float* __restrict__ out,
    const float* __restrict__ g, const float* __restrict__ b)
{
    int row = blockIdx.x;
    const float* p = in + (long long)row * 768;
    float s = 0.f, s2 = 0.f;
    for (int i = threadIdx.x; i < 768; i += 256) { float v = p[i]; s += v; s2 += v * v; }
    int lane = threadIdx.x & 31, wave = threadIdx.x >> 5;
#pragma unroll
    for (int o = 16; o > 0; o >>= 1) { s += __shfl_xor(s, o, 32); s2 += __shfl_xor(s2, o, 32); }
    __shared__ float rs[8], rq[8], fm, fv;
    if (lane == 0) { rs[wave] = s; rq[wave] = s2; }
    __syncthreads();
    if (wave == 0) {
        s  = (lane < 8) ? rs[lane] : 0.f;
        s2 = (lane < 8) ? rq[lane] : 0.f;
#pragma unroll
        for (int o = 4; o > 0; o >>= 1) { s += __shfl_xor(s, o, 32); s2 += __shfl_xor(s2, o, 32); }
        if (lane == 0) {
            float m = s * (1.0f / 768.0f);
            fm = m;
            fv = s2 * (1.0f / 768.0f) - m * m;
        }
    }
    __syncthreads();
    float mean = fm, inv = rsqrtf(fv + 1e-5f);
    for (int i = threadIdx.x; i < 768; i += 256)
        out[(long long)row * 768 + i] = (p[i] - mean) * inv * g[i] + b[i];
}

extern "C" void kernel_launch(void* const* d_in, const int* in_sizes, int n_in,
                              void* d_out, int out_size, void* d_ws, size_t ws_size,
                              hipStream_t stream)
{
    const float* x    = (const float*)d_in[0];
    const unsigned char* mask = (const unsigned char*)d_in[1];
    const float* wqkv = (const float*)d_in[2];
    const float* bqkv = (const float*)d_in[3];
    const float* wo   = (const float*)d_in[4];
    const float* bo   = (const float*)d_in[5];
    const float* w1   = (const float*)d_in[6];
    const float* b1   = (const float*)d_in[7];
    const float* w2   = (const float*)d_in[8];
    const float* b2   = (const float*)d_in[9];
    const float* ln1g = (const float*)d_in[10];
    const float* ln1b = (const float*)d_in[11];
    const float* ln2g = (const float*)d_in[12];
    const float* ln2b = (const float*)d_in[13];

    constexpr int Bb = 16, S = 512, H = 768, NH = 12, HD = 64;
    constexpr int M = Bb * S;  // 8192

    float* ws   = (float*)d_ws;
    float* qkv  = ws;                        // [M, 3H]
    float* acat = qkv  + (size_t)M * 3 * H;  // [M, H]  attn heads concatenated
    float* res1 = acat + (size_t)M * H;      // [M, H]  x + attn_proj
    float* x1   = res1 + (size_t)M * H;      // [M, H]  LN1 output
    float* h1   = x1   + (size_t)M * H;      // [M, 4H] GELU(ffn1)
    float* res2 = h1   + (size_t)M * 4 * H;  // [M, H]  x1 + ffn2

    float* xout = (float*)d_out;             // [M, H]
    float* attn = xout + (size_t)M * H;      // [B, NH, S, S]

    dim3 blk(256);

    // 1) qkv = x @ Wqkv^T + b   (q cols scaled by 0.125); exact tiles -> no guards
    gemm_bf16_wmma<EPI_BIAS_QSCALE, true, false><<<dim3(18, 64, 1), blk, 0, stream>>>(
        x, H, 0, 0,  wqkv, H, 0, 0,  qkv, 3 * H, 0, 0,
        M, 3 * H, H, 1, bqkv, nullptr, 0, nullptr);

    // 2) scores = q @ k^T per (b,h), masked, written into attn region of d_out
    gemm_bf16_wmma<EPI_MASK, true, false><<<dim3(4, 4, Bb * NH), blk, 0, stream>>>(
        qkv,     3 * H, (long long)S * 3 * H, HD,
        qkv + H, 3 * H, (long long)S * 3 * H, HD,
        attn, S, (long long)NH * S * S, (long long)S * S,
        S, S, HD, NH, nullptr, nullptr, 0, mask);

    // 3) row softmax in place (attn_weights output)
    softmax512<<<dim3((Bb * NH * S) / 8), blk, 0, stream>>>(attn, Bb * NH * S);

    // 4) attn_out = attn @ v  -> acat [M, H]   (N=64 < BN: needs guards)
    gemm_bf16_wmma<EPI_NONE, false, true><<<dim3(1, 4, Bb * NH), blk, 0, stream>>>(
        attn, S, (long long)NH * S * S, (long long)S * S,
        qkv + 2 * H, 3 * H, (long long)S * 3 * H, HD,
        acat, H, (long long)S * H, HD,
        S, HD, S, NH, nullptr, nullptr, 0, nullptr);

    // 5) out-proj + bias + residual(x) -> res1
    gemm_bf16_wmma<EPI_BIAS_RES, true, false><<<dim3(6, 64, 1), blk, 0, stream>>>(
        acat, H, 0, 0,  wo, H, 0, 0,  res1, H, 0, 0,
        M, H, H, 1, bo, x, H, nullptr);

    // 6) LN1 -> x1
    layernorm768<<<dim3(M), blk, 0, stream>>>(res1, x1, ln1g, ln1b);

    // 7) ffn1: GELU(x1 @ W1^T + b1) -> h1
    gemm_bf16_wmma<EPI_BIAS_GELU, true, false><<<dim3(24, 64, 1), blk, 0, stream>>>(
        x1, H, 0, 0,  w1, H, 0, 0,  h1, 4 * H, 0, 0,
        M, 4 * H, H, 1, b1, nullptr, 0, nullptr);

    // 8) ffn2 + bias + residual(x1) -> res2
    gemm_bf16_wmma<EPI_BIAS_RES, true, false><<<dim3(6, 64, 1), blk, 0, stream>>>(
        h1, 4 * H, 0, 0,  w2, 4 * H, 0, 0,  res2, H, 0, 0,
        M, H, 4 * H, 1, b2, x1, H, nullptr);

    // 9) LN2 -> x output
    layernorm768<<<dim3(M), blk, 0, stream>>>(res2, xout, ln2g, ln2b);
}
